// GAT_Hetersum_19971597926650
// MI455X (gfx1250) — matrix operand order repaired and checked by reading, not compiled, not run on hardware
//
#include <hip/hip_runtime.h>

typedef __bf16 bf16_t;
typedef __attribute__((ext_vector_type(16))) __bf16 v16bf;
typedef __attribute__((ext_vector_type(8)))  float  v8f;

#define NN   20000
#define NE   400000
#define NHID 512
#define NH   8
#define HD   64
#define FFH  2048
#define NSENT 2000
#define MCHUNK 4096

// ---------------- WMMA GEMM: C[M,N] = A[M,K](bf16) x Bsw(bf16, fragment-swizzled) ----
// block tile 64x64, K-step 64, 8 waves (2x4), each wave -> two 16x16 f32 accumulators.
// A staged in LDS (double-buffered, async global->LDS copies); B fragments loaded
// directly from a pre-swizzled global layout:
//   Bsw[(((ks*(N/16)) + nt)*32 + lane)*16 + j] = B[ks*32 + (lane/16)*16 + j][nt*16 + (lane%16)]
#define TM 64
#define TN 64
#define TK 64
#define AP 72            // LDS row pitch in bf16 units (144B rows, 16B aligned)
#define ABUF (TM * AP)   // one A buffer in bf16 elems

__device__ __forceinline__ void stage_async_b256(unsigned ldsByte, const bf16_t* gp) {
    unsigned long long ga = (unsigned long long)(uintptr_t)gp;
    asm volatile("global_load_async_to_lds_b128 %0, %1, off\n\t"
                 "global_load_async_to_lds_b128 %0, %1, off offset:16"
                 :: "v"(ldsByte), "v"(ga) : "memory");
}

__global__ __launch_bounds__(256)
void wmma_gemm_bf16(const bf16_t* __restrict__ A, const bf16_t* __restrict__ Bsw,
                    int M, int N, int K,
                    const float* __restrict__ bias,
                    const float* __restrict__ residual, int do_relu,
                    float* __restrict__ Cf, bf16_t* __restrict__ Cb)
{
    __shared__ __align__(16) bf16_t Asm[2 * ABUF];

    const int tid  = threadIdx.x;
    const int lane = tid & 31;
    const int wave = tid >> 5;          // 0..7
    const int wm   = wave >> 2;         // 0..1  (M sub-block)
    const int wn   = wave & 3;          // 0..3  (N sub-block)
    const int lq   = lane >> 4;         // lane half 0/1
    const int ln   = lane & 15;

    const int bm = blockIdx.y * TM;
    const int bn = blockIdx.x * TN;

    // A staging coords (per thread: 32 contiguous bytes of one row)
    const int srow = tid >> 2;              // 0..63
    const int sseg = (tid & 3) * 16;        // 0,16,32,48 (bf16 elems)
    const int grow = bm + srow;
    const bool fullM = (bm + TM <= M);
    const unsigned lb0 = (unsigned)(uintptr_t)&Asm[srow * AP + sseg];

    // B fragment source (per wave: one 16-col tile; per lane: 32B contiguous)
    const int nt = (bn >> 4) + wn;
    const size_t bks_stride = (size_t)(N >> 4) * 32 * 16;   // elems per 32-wide K step
    const bf16_t* bbase = Bsw + ((size_t)nt * 32 + lane) * 16;

    v8f acc0 = {};
    v8f acc1 = {};
    union Frag { uint4 u[2]; v16bf v; };

    // ---- stage first tile ----
    {
        if (fullM) {
            stage_async_b256(lb0, A + (size_t)grow * K + sseg);
        } else {
            uint4 v0 = make_uint4(0u,0u,0u,0u), v1 = v0;
            if (grow < M) {
                const uint4* s = (const uint4*)(A + (size_t)grow * K + sseg);
                v0 = s[0]; v1 = s[1];
            }
            uint4* d = (uint4*)&Asm[srow * AP + sseg];
            d[0] = v0; d[1] = v1;
        }
        asm volatile("s_wait_asynccnt 0x0" ::: "memory");
        __syncthreads();
    }

    const int niter = K / TK;
    for (int it = 0; it < niter; ++it) {
        const int k0 = it * TK;
        const int cur = it & 1;
        // prefetch next tile into the other buffer (async; overlapped with WMMA below)
        if (it + 1 < niter) {
            const int nxt = cur ^ 1;
            if (fullM) {
                stage_async_b256(lb0 + (unsigned)(nxt * ABUF * 2),
                                 A + (size_t)grow * K + (k0 + TK) + sseg);
            } else {
                uint4 v0 = make_uint4(0u,0u,0u,0u), v1 = v0;
                if (grow < M) {
                    const uint4* s = (const uint4*)(A + (size_t)grow * K + (k0 + TK) + sseg);
                    v0 = s[0]; v1 = s[1];
                }
                uint4* d = (uint4*)&Asm[nxt * ABUF + srow * AP + sseg];
                d[0] = v0; d[1] = v1;
            }
        }

        const int abase = cur * ABUF;
        #pragma unroll
        for (int kk = 0; kk < TK; kk += 32) {
            Frag a0, a1, b;
            // A fragment: lane ln = row m; K elems {kA..kA+15}
            const int kA  = kk + lq * 16;
            const int ar0 = abase + (wm * 16 + ln) * AP;
            const int ar1 = ar0 + 32 * AP;
            a0.u[0] = *(const uint4*)&Asm[ar0 + kA];
            a0.u[1] = *(const uint4*)&Asm[ar0 + kA + 8];
            a1.u[0] = *(const uint4*)&Asm[ar1 + kA];
            a1.u[1] = *(const uint4*)&Asm[ar1 + kA + 8];
            // B fragment: 32B contiguous from the swizzled panel
            const int ks = (k0 + kk) >> 5;
            const uint4* bp = (const uint4*)(bbase + (size_t)ks * bks_stride);
            b.u[0] = bp[0];
            b.u[1] = bp[1];

            acc0 = __builtin_amdgcn_wmma_f32_16x16x32_bf16(false, a0.v, false, b.v,
                                                           (short)0, acc0, false, false);
            acc1 = __builtin_amdgcn_wmma_f32_16x16x32_bf16(false, a1.v, false, b.v,
                                                           (short)0, acc1, false, false);
        }
        asm volatile("s_wait_asynccnt 0x0" ::: "memory");
        __syncthreads();
    }

    // ---- epilogue: lane ln = col; VGPR r -> row r + 8*lq ----
    const int col = bn + wn * 16 + ln;
    #pragma unroll
    for (int r = 0; r < 8; ++r) {
        const int rt = r + lq * 8;
        const int g0 = bm + wm * 16 + rt;
        const int g1 = g0 + 32;
        float v0 = acc0[r];
        float v1 = acc1[r];
        if (bias) { float bv = bias[col]; v0 += bv; v1 += bv; }
        if (do_relu) { v0 = fmaxf(v0, 0.f); v1 = fmaxf(v1, 0.f); }
        if (g0 < M) {
            float o = v0;
            if (residual) o += residual[(size_t)g0 * N + col];
            if (Cf) Cf[(size_t)g0 * N + col] = o;
            if (Cb) Cb[(size_t)g0 * N + col] = (bf16_t)o;
        }
        if (g1 < M) {
            float o = v1;
            if (residual) o += residual[(size_t)g1 * N + col];
            if (Cf) Cf[(size_t)g1 * N + col] = o;
            if (Cb) Cb[(size_t)g1 * N + col] = (bf16_t)o;
        }
    }
}

// ---------------- weight repacks into fragment-swizzled bf16 panels ----------------
// element i = (((ks*(N/16)) + nt)*32 + lane)*16 + j  maps to  B[k][n] with
//   k = ks*32 + (lane/16)*16 + j ;  n = nt*16 + (lane%16)
__global__ void repack_Wz_kernel(const float* __restrict__ Wh, bf16_t* __restrict__ Bsw) {
    int i = blockIdx.x * blockDim.x + threadIdx.x;
    if (i >= NHID * NHID) return;
    int j = i & 15, lane = (i >> 4) & 31, t = i >> 9;
    int ntile = t & 31, ks = t >> 5;                       // N/16 = 32
    int k = ks * 32 + (lane >> 4) * 16 + j;
    int n = ntile * 16 + (lane & 15);
    int h = n >> 6, d = n & 63;
    Bsw[i] = (bf16_t)Wh[((size_t)h * NHID + k) * HD + d];
}
__global__ void repack_w1T_kernel(const float* __restrict__ w1, bf16_t* __restrict__ Bsw) {
    int i = blockIdx.x * blockDim.x + threadIdx.x;
    if (i >= NHID * FFH) return;
    int j = i & 15, lane = (i >> 4) & 31, t = i >> 9;
    int ntile = t & 127, ks = t >> 7;                      // N/16 = 128
    int k = ks * 32 + (lane >> 4) * 16 + j;
    int n = ntile * 16 + (lane & 15);
    Bsw[i] = (bf16_t)w1[(size_t)n * NHID + k];             // B[k][n] = w1[n][k]
}
__global__ void repack_w2T_kernel(const float* __restrict__ w2, bf16_t* __restrict__ Bsw) {
    int i = blockIdx.x * blockDim.x + threadIdx.x;
    if (i >= FFH * NHID) return;
    int j = i & 15, lane = (i >> 4) & 31, t = i >> 9;
    int ntile = t & 31, ks = t >> 5;                       // N/16 = 32
    int k = ks * 32 + (lane >> 4) * 16 + j;
    int n = ntile * 16 + (lane & 15);
    Bsw[i] = (bf16_t)w2[(size_t)n * FFH + k];              // B[k][n] = w2[n][k]
}

// ---------------- elementwise / edge kernels ----------------
__global__ void init_h_kernel(const float* __restrict__ f, float* __restrict__ hcur,
                              bf16_t* __restrict__ hb, int n) {
    int i = blockIdx.x * blockDim.x + threadIdx.x;
    if (i >= n) return;
    float v = f[i];
    hcur[i] = v;
    hb[i] = (bf16_t)v;
}

__global__ void zero_f32_kernel(float* p, int n) {
    int i = blockIdx.x * blockDim.x + threadIdx.x;
    if (i < n) p[i] = 0.f;
}
__global__ void zero_u32_kernel(unsigned* p, int n) {
    int i = blockIdx.x * blockDim.x + threadIdx.x;
    if (i < n) p[i] = 0u;
}

__global__ void attn_scores_kernel(const float* __restrict__ z, const float* __restrict__ al,
                                   const float* __restrict__ ar,
                                   float* __restrict__ ssrc, float* __restrict__ sdst, int total) {
    int i = blockIdx.x * blockDim.x + threadIdx.x;       // node*8 + h
    if (i >= total) return;
    int node = i >> 3, h = i & 7;
    const float* zp = z + (size_t)node * NHID + h * HD;
    const float* ap = al + h * HD;
    const float* bp = ar + h * HD;
    float a = 0.f, b = 0.f;
    #pragma unroll 8
    for (int d = 0; d < HD; ++d) { float v = zp[d]; a += v * ap[d]; b += v * bp[d]; }
    ssrc[i] = a; sdst[i] = b;
}

__global__ void edge_logit_max_kernel(const int* __restrict__ src, const int* __restrict__ dst,
                                      const float* __restrict__ ssrc, const float* __restrict__ sdst,
                                      float* __restrict__ ebuf, unsigned* __restrict__ nmaxu, int total) {
    int i = blockIdx.x * blockDim.x + threadIdx.x;       // e*8 + h
    if (i >= total) return;
    int e = i >> 3, h = i & 7;
    int s = src[e], d = dst[e];
    float x = ssrc[s * 8 + h] + sdst[d * 8 + h];
    x = x > 0.f ? x : 0.01f * x;                          // leaky_relu 0.01
    ebuf[i] = x;
    unsigned bits = __float_as_uint(x);
    unsigned key  = bits ^ (unsigned)(((int)bits >> 31) | (int)0x80000000);  // order-preserving map
    atomicMax(&nmaxu[d * 8 + h], key);
}

__global__ void edge_expsum_kernel(const int* __restrict__ dst, const unsigned* __restrict__ nmaxu,
                                   float* __restrict__ ebuf, float* __restrict__ nsum, int total) {
    int i = blockIdx.x * blockDim.x + threadIdx.x;
    if (i >= total) return;
    int e = i >> 3, h = i & 7;
    int d = dst[e];
    unsigned u = nmaxu[d * 8 + h];
    unsigned mask = (u >> 31) ? 0x80000000u : 0xFFFFFFFFu;
    float m = __uint_as_float(u ^ mask);
    float ex = __expf(ebuf[i] - m);
    ebuf[i] = ex;
    atomicAdd(&nsum[d * 8 + h], ex);
}

__global__ void edge_scatter_kernel(const int* __restrict__ src, const int* __restrict__ dst,
                                    const float* __restrict__ ebuf, const float* __restrict__ nsum,
                                    const float* __restrict__ z, float* __restrict__ sh, int total) {
    int i = blockIdx.x * blockDim.x + threadIdx.x;       // e*512 + c
    if (i >= total) return;
    int e = i >> 9, c = i & 511;
    int h = c >> 6;
    int s = src[e], d = dst[e];
    float alpha = ebuf[e * 8 + h] / nsum[d * 8 + h];
    atomicAdd(&sh[(size_t)d * NHID + c], alpha * z[(size_t)s * NHID + c]);
}

__global__ void elu_residual_kernel(const float* __restrict__ sh, float* __restrict__ hcur,
                                    bf16_t* __restrict__ hb, int n) {
    int i = blockIdx.x * blockDim.x + threadIdx.x;
    if (i >= n) return;
    float x = sh[i];
    float e = x > 0.f ? x : (__expf(x) - 1.f);
    float v = e + hcur[i];
    hcur[i] = v;
    hb[i] = (bf16_t)v;
}

__global__ __launch_bounds__(256)
void layernorm_kernel(const float* __restrict__ y, const float* __restrict__ g,
                      const float* __restrict__ b, float* __restrict__ hcur,
                      bf16_t* __restrict__ hb) {
    __shared__ float red[256];
    const int row = blockIdx.x;
    const int tid = threadIdx.x;
    const float* yr = y + (size_t)row * NHID;
    float s = 0.f;
    for (int c = tid; c < NHID; c += 256) s += yr[c];
    red[tid] = s; __syncthreads();
    for (int o = 128; o > 0; o >>= 1) { if (tid < o) red[tid] += red[tid + o]; __syncthreads(); }
    float mu = red[0] * (1.f / NHID);
    __syncthreads();
    float vs = 0.f;
    for (int c = tid; c < NHID; c += 256) { float dlt = yr[c] - mu; vs += dlt * dlt; }
    red[tid] = vs; __syncthreads();
    for (int o = 128; o > 0; o >>= 1) { if (tid < o) red[tid] += red[tid + o]; __syncthreads(); }
    float rstd = rsqrtf(red[0] * (1.f / NHID) + 1e-5f);
    for (int c = tid; c < NHID; c += 256) {
        float o = (yr[c] - mu) * rstd * g[c] + b[c];
        hcur[(size_t)row * NHID + c] = o;
        hb[(size_t)row * NHID + c] = (bf16_t)o;
    }
}

__global__ void gather_rows_kernel(const float* __restrict__ h, const int* __restrict__ sidx,
                                   float* __restrict__ out, int total) {
    int i = blockIdx.x * blockDim.x + threadIdx.x;       // s*512 + c
    if (i >= total) return;
    int s = i >> 9, c = i & 511;
    out[i] = h[(size_t)sidx[s] * NHID + c];
}

// ---------------- host orchestration ----------------
static inline int cdiv(int a, int b) { return (a + b - 1) / b; }

extern "C" void kernel_launch(void* const* d_in, const int* in_sizes, int n_in,
                              void* d_out, int out_size, void* d_ws, size_t ws_size,
                              hipStream_t stream) {
    const float* feats = (const float*)d_in[0];
    const float* Wh    = (const float*)d_in[1];
    const float* al    = (const float*)d_in[2];
    const float* ar    = (const float*)d_in[3];
    const float* w1    = (const float*)d_in[4];
    const float* b1    = (const float*)d_in[5];
    const float* w2    = (const float*)d_in[6];
    const float* b2    = (const float*)d_in[7];
    const float* ln_g  = (const float*)d_in[8];
    const float* ln_b  = (const float*)d_in[9];
    const int* esrc    = (const int*)d_in[10];
    const int* edst    = (const int*)d_in[11];
    const int* sidx    = (const int*)d_in[12];
    float* out = (float*)d_out;

    // carve workspace (256B aligned slices)
    size_t off = 0;
    char* base = (char*)d_ws;
    auto carve = [&](size_t bytes) -> void* {
        void* r = base + off;
        off += (bytes + 255) & ~(size_t)255;
        return r;
    };
    bf16_t*   hb    = (bf16_t*)  carve((size_t)NN * NHID * 2);
    float*    hcur  = (float*)   carve((size_t)NN * NHID * 4);
    float*    zbuf  = (float*)   carve((size_t)NN * NHID * 4);   // z, later reused as y2
    float*    shb   = (float*)   carve((size_t)NN * NHID * 4);
    bf16_t*   y1    = (bf16_t*)  carve((size_t)MCHUNK * FFH * 2);
    float*    ssrc  = (float*)   carve((size_t)NN * NH * 4);
    float*    sdst  = (float*)   carve((size_t)NN * NH * 4);
    unsigned* nmaxu = (unsigned*)carve((size_t)NN * NH * 4);
    float*    nsum  = (float*)   carve((size_t)NN * NH * 4);
    float*    ebuf  = (float*)   carve((size_t)NE * NH * 4);
    bf16_t*   Bz    = (bf16_t*)  carve((size_t)NHID * NHID * 2);
    bf16_t*   Bw1   = (bf16_t*)  carve((size_t)NHID * FFH * 2);
    bf16_t*   Bw2   = (bf16_t*)  carve((size_t)FFH * NHID * 2);
    (void)ws_size; (void)n_in; (void)in_sizes; (void)out_size;

    const int TPB = 256;
    const int nh_elems = NN * NHID;                 // 10,240,000
    const int nscore   = NN * NH;                   // 160,000
    const int nedgeh   = NE * NH;                   // 3,200,000
    const int nscat    = NE * NHID;                 // 204,800,000

    // weight repacks (fp32 -> bf16, fragment-swizzled panels)
    repack_w1T_kernel<<<cdiv(NHID * FFH, TPB), TPB, 0, stream>>>(w1, Bw1);
    repack_w2T_kernel<<<cdiv(FFH * NHID, TPB), TPB, 0, stream>>>(w2, Bw2);

    // h = feats (fp32 + bf16 copies)
    init_h_kernel<<<cdiv(nh_elems, TPB), TPB, 0, stream>>>(feats, hcur, hb, nh_elems);

    for (int step = 0; step < 2; ++step) {
        const float* Wh_s = Wh + (size_t)step * NH * NHID * HD;
        const float* al_s = al + (size_t)step * NH * HD;
        const float* ar_s = ar + (size_t)step * NH * HD;

        // z = h x W (all heads concatenated): [20000,512] x [512,512]
        repack_Wz_kernel<<<cdiv(NHID * NHID, TPB), TPB, 0, stream>>>(Wh_s, Bz);
        {
            dim3 grid(NHID / TN, cdiv(NN, TM));
            wmma_gemm_bf16<<<grid, 256, 0, stream>>>(hb, Bz, NN, NHID, NHID,
                                                     nullptr, nullptr, 0, zbuf, nullptr);
        }

        // attention scores, segment softmax, message scatter
        attn_scores_kernel<<<cdiv(nscore, TPB), TPB, 0, stream>>>(zbuf, al_s, ar_s, ssrc, sdst, nscore);
        zero_u32_kernel<<<cdiv(nscore, TPB), TPB, 0, stream>>>(nmaxu, nscore);
        zero_f32_kernel<<<cdiv(nscore, TPB), TPB, 0, stream>>>(nsum, nscore);
        zero_f32_kernel<<<cdiv(nh_elems, TPB), TPB, 0, stream>>>(shb, nh_elems);
        edge_logit_max_kernel<<<cdiv(nedgeh, TPB), TPB, 0, stream>>>(esrc, edst, ssrc, sdst, ebuf, nmaxu, nedgeh);
        edge_expsum_kernel<<<cdiv(nedgeh, TPB), TPB, 0, stream>>>(edst, nmaxu, ebuf, nsum, nedgeh);
        edge_scatter_kernel<<<cdiv(nscat, TPB), TPB, 0, stream>>>(esrc, edst, ebuf, nsum, zbuf, shb, nscat);

        // h = elu(sh) + h
        elu_residual_kernel<<<cdiv(nh_elems, TPB), TPB, 0, stream>>>(shb, hcur, hb, nh_elems);

        // FFN (chunked over rows): y2 = relu(h x w1^T + b1) x w2^T + b2 + h  -> zbuf
        for (int mc = 0; mc < NN; mc += MCHUNK) {
            int rows = NN - mc < MCHUNK ? NN - mc : MCHUNK;
            dim3 g1(FFH / TN, cdiv(rows, TM));
            wmma_gemm_bf16<<<g1, 256, 0, stream>>>(hb + (size_t)mc * NHID, Bw1,
                                                   rows, FFH, NHID,
                                                   b1, nullptr, 1, nullptr, y1);
            dim3 g2(NHID / TN, cdiv(rows, TM));
            wmma_gemm_bf16<<<g2, 256, 0, stream>>>(y1, Bw2,
                                                   rows, NHID, FFH,
                                                   b2, hcur + (size_t)mc * NHID, 0,
                                                   zbuf + (size_t)mc * NHID, nullptr);
        }

        // h = LayerNorm(y2)
        layernorm_kernel<<<NN, 256, 0, stream>>>(zbuf, ln_g, ln_b, hcur, hb);
    }

    // out = h[sent_idx]
    gather_rows_kernel<<<cdiv(NSENT * NHID, TPB), TPB, 0, stream>>>(hcur, sidx, out, NSENT * NHID);
}
